// _GraphNonLocal_1821066134159
// MI455X (gfx1250) — matrix operand order repaired and checked by reading, not compiled, not run on hardware
//
#include <hip/hip_runtime.h>
#include <hip/hip_bf16.h>

typedef __attribute__((ext_vector_type(16))) _Float16 v16h;
typedef __attribute__((ext_vector_type(8)))  _Float16 h8;
typedef __attribute__((ext_vector_type(8)))  float    v8f;

namespace {
constexpr int kB = 256, kN = 1024, kC = 512, kD = 256, kM = 256;
constexpr int XS = 520;   // LDS stride (halves) for xg strip, keeps 16B alignment
constexpr int SS = 264;   // LDS stride for scores(f32) / att(f16) / y(f16)
constexpr float kEps = 1e-5f;
}

// Build a 16-half WMMA operand from two 16-byte chunks (A: gap 16 halves, B: gap 16).
__device__ __forceinline__ v16h ldpair(const _Float16* p, const _Float16* q) {
  h8 lo = *(const h8*)p;
  h8 hi = *(const h8*)q;
  return __builtin_shufflevector(lo, hi, 0,1,2,3,4,5,6,7,8,9,10,11,12,13,14,15);
}

__device__ __forceinline__ v8f wmma16(v16h a, v16h b, v8f c) {
  return __builtin_amdgcn_wmma_f32_16x16x32_f16(false, a, false, b, (short)0, c,
                                                false, false);
}

// ---------------------------------------------------------------------------
// Kernel W: f32 -> f16 weight conversion (theta_w, g_w, phi_w all DxC; W_w CxD)
// ---------------------------------------------------------------------------
__global__ void __launch_bounds__(256)
k_wcvt(const float* __restrict__ tw, const float* __restrict__ gw,
       const float* __restrict__ pw, const float* __restrict__ Ww,
       _Float16* __restrict__ tw16, _Float16* __restrict__ gw16,
       _Float16* __restrict__ pw16, _Float16* __restrict__ Ww16) {
  int i = blockIdx.x * 256 + threadIdx.x;
  if (i < kD * kC) {
    tw16[i] = (_Float16)tw[i];
    gw16[i] = (_Float16)gw[i];
    pw16[i] = (_Float16)pw[i];
    Ww16[i] = (_Float16)Ww[i];
  }
}

// ---------------------------------------------------------------------------
// Kernel P: gather + theta/g/phi projections (+ SUB=4 maxpool for g, phi).
// One wave = one 16-row strip of grouped x. 2 waves / 64-thread block.
// ---------------------------------------------------------------------------
__global__ void __launch_bounds__(64)
k_proj(const float* __restrict__ x, const int* __restrict__ perm,
       const _Float16* __restrict__ tw16, const _Float16* __restrict__ gw16,
       const _Float16* __restrict__ pw16,
       const float* __restrict__ theta_b, const float* __restrict__ g_b,
       const float* __restrict__ phi_b,
       _Float16* __restrict__ theta16, _Float16* __restrict__ phi16,
       _Float16* __restrict__ gT16) {
  __shared__ _Float16 xg[2][16][XS];
  __shared__ int permRow[2][16];
  const int w = threadIdx.x >> 5, lane = threadIdx.x & 31;
  const int strip = blockIdx.x * 2 + w;
  const int b = strip >> 6;
  const int n0 = (strip & 63) << 4;
  const int lr = lane & 15, lh = lane >> 4;

  if (lane < 16) permRow[w][lane] = perm[n0 + lane];
  __syncthreads();

  // Gather 16 rows x 512 cols of x (f32) -> f16 LDS strip.
  for (int i = lane; i < 16 * 128; i += 32) {
    int r = i >> 7, cq = i & 127;
    const float4 f4 = ((const float4*)(x + ((size_t)b * kN + permRow[w][r]) * kC))[cq];
    _Float16* d = &xg[w][r][cq * 4];
    d[0] = (_Float16)f4.x; d[1] = (_Float16)f4.y;
    d[2] = (_Float16)f4.z; d[3] = (_Float16)f4.w;
  }
  __syncthreads();

  // theta: 16 d-tiles, K = C = 512 -> 16 chained wmma per tile.
#pragma unroll 1
  for (int dt = 0; dt < 16; ++dt) {
    v8f acc = {};
#pragma unroll
    for (int kc = 0; kc < 16; ++kc) {
      const _Float16* ap = &xg[w][lr][kc * 32 + lh * 8];
      const _Float16* bp = tw16 + (size_t)(dt * 16 + lr) * kC + kc * 32 + lh * 8;
      acc = wmma16(ldpair(ap, ap + 16), ldpair(bp, bp + 16), acc);
    }
    const int col = dt * 16 + lr;
    const float bias = theta_b[col];
#pragma unroll
    for (int v = 0; v < 8; ++v)
      theta16[((size_t)b * kN + n0 + lh * 8 + v) * kD + col] = (_Float16)(acc[v] + bias);
  }

  // g (stored transposed [d][m]) and phi ([m][d]), pooled 4:1 along n.
#pragma unroll 1
  for (int which = 0; which < 2; ++which) {
    const _Float16* wm = which ? pw16 : gw16;
    const float* bb = which ? phi_b : g_b;
#pragma unroll 1
    for (int dt = 0; dt < 16; ++dt) {
      v8f acc = {};
#pragma unroll
      for (int kc = 0; kc < 16; ++kc) {
        const _Float16* ap = &xg[w][lr][kc * 32 + lh * 8];
        const _Float16* bp = wm + (size_t)(dt * 16 + lr) * kC + kc * 32 + lh * 8;
        acc = wmma16(ldpair(ap, ap + 16), ldpair(bp, bp + 16), acc);
      }
      const int col = dt * 16 + lr;
      const float bias = bb[col];
      // C layout: this lane holds rows lh*8 .. lh*8+7 of the tile in acc[0..7].
      float p0 = fmaxf(fmaxf(acc[0], acc[1]), fmaxf(acc[2], acc[3])) + bias;
      float p1 = fmaxf(fmaxf(acc[4], acc[5]), fmaxf(acc[6], acc[7])) + bias;
      const int m0 = (n0 >> 2) + lh * 2;
      if (which == 0) {
        gT16[((size_t)b * kD + col) * kM + m0]     = (_Float16)p0;
        gT16[((size_t)b * kD + col) * kM + m0 + 1] = (_Float16)p1;
      } else {
        phi16[((size_t)b * kM + m0) * kD + col]     = (_Float16)p0;
        phi16[((size_t)b * kM + m0 + 1) * kD + col] = (_Float16)p1;
      }
    }
  }
}

// ---------------------------------------------------------------------------
// Kernel A: scores -> softmax -> y = att*g -> wy = y*W^T -> BN+residual+scatter
// One wave = one 16-row n-strip. Per-wave LDS does C-layout -> A-layout turns.
// ---------------------------------------------------------------------------
__global__ void __launch_bounds__(64)
k_attn(const float* __restrict__ x, const int* __restrict__ perm,
       const _Float16* __restrict__ theta16, const _Float16* __restrict__ phi16,
       const _Float16* __restrict__ gT16, const _Float16* __restrict__ Ww16,
       const float* __restrict__ W_b,
       const float* __restrict__ bn_gamma, const float* __restrict__ bn_beta,
       const float* __restrict__ bn_mean, const float* __restrict__ bn_var,
       float* __restrict__ out) {
  __shared__ float sc[2][16][SS];          // scores f32; later reused as y16
  __shared__ _Float16 attb[2][16][SS];     // attention weights f16
  __shared__ int permRow[2][16];
  const int w = threadIdx.x >> 5, lane = threadIdx.x & 31;
  const int strip = blockIdx.x * 2 + w;
  const int b = strip >> 6;
  const int n0 = (strip & 63) << 4;
  const int lr = lane & 15, lh = lane >> 4;

  if (lane < 16) permRow[w][lane] = perm[n0 + lane];

  // ---- scores S = theta · phi^T : 16 x 256, K = D = 256 ----
  v16h aTh[8];
#pragma unroll
  for (int kc = 0; kc < 8; ++kc) {
    const _Float16* ap = theta16 + ((size_t)b * kN + n0 + lr) * kD + kc * 32 + lh * 8;
    aTh[kc] = ldpair(ap, ap + 16);
  }
#pragma unroll 1
  for (int mt = 0; mt < 16; ++mt) {
    if (mt < 15)
      __builtin_prefetch(phi16 + ((size_t)b * kM + (mt + 1) * 16 + lr) * kD);
    v8f acc = {};
#pragma unroll
    for (int kc = 0; kc < 8; ++kc) {
      const _Float16* bp = phi16 + ((size_t)b * kM + mt * 16 + lr) * kD + kc * 32 + lh * 8;
      acc = wmma16(aTh[kc], ldpair(bp, bp + 16), acc);
    }
#pragma unroll
    for (int v = 0; v < 8; ++v) sc[w][lh * 8 + v][mt * 16 + lr] = acc[v];
  }
  __syncthreads();

  // ---- row softmax over m = 256 (2 lanes per row, half-wave shuffles) ----
  {
    const float* prow = &sc[w][lr][lh * 128];
    _Float16* arow = &attb[w][lr][lh * 128];
    float mx = -3.402823466e38f;
#pragma unroll 4
    for (int i = 0; i < 128; ++i) mx = fmaxf(mx, prow[i]);
    mx = fmaxf(mx, __shfl_xor(mx, 16, 32));
    float s = 0.f;
#pragma unroll 4
    for (int i = 0; i < 128; ++i) {
      float e = __expf(prow[i] - mx);
      s += e;
      arow[i] = (_Float16)e;
    }
    s += __shfl_xor(s, 16, 32);
    const float inv = 1.0f / s;
#pragma unroll 4
    for (int i = 0; i < 128; ++i) arow[i] = (_Float16)((float)arow[i] * inv);
  }
  __syncthreads();

  // ---- y = att · g : 16 x 256, K = M = 256 (gT provides B columns) ----
  _Float16* y16 = (_Float16*)&sc[w][0][0];  // reuse score buffer (f16, stride SS)
  v16h aAt[8];
#pragma unroll
  for (int kc = 0; kc < 8; ++kc) {
    const _Float16* ap = &attb[w][lr][kc * 32 + lh * 8];
    aAt[kc] = ldpair(ap, ap + 16);
  }
#pragma unroll 1
  for (int dt = 0; dt < 16; ++dt) {
    v8f acc = {};
#pragma unroll
    for (int kc = 0; kc < 8; ++kc) {
      const _Float16* bp = gT16 + ((size_t)b * kD + dt * 16 + lr) * kM + kc * 32 + lh * 8;
      acc = wmma16(aAt[kc], ldpair(bp, bp + 16), acc);
    }
#pragma unroll
    for (int v = 0; v < 8; ++v)
      y16[(size_t)(lh * 8 + v) * SS + dt * 16 + lr] = (_Float16)acc[v];
  }
  __syncthreads();

  // ---- wy = y · W^T (16 x 512, K = D = 256) + bias + BN + residual ----
  v16h aY[8];
#pragma unroll
  for (int kc = 0; kc < 8; ++kc) {
    const _Float16* ap = &y16[(size_t)lr * SS + kc * 32 + lh * 8];
    aY[kc] = ldpair(ap, ap + 16);
  }
#pragma unroll 1
  for (int ct = 0; ct < 32; ++ct) {
    v8f acc = {};
#pragma unroll
    for (int kc = 0; kc < 8; ++kc) {
      const _Float16* bp = Ww16 + (size_t)(ct * 16 + lr) * kD + kc * 32 + lh * 8;
      acc = wmma16(aY[kc], ldpair(bp, bp + 16), acc);
    }
    const int c = ct * 16 + lr;
    const float wb = W_b[c];
    const float scale = bn_gamma[c] * rsqrtf(bn_var[c] + kEps);
    const float shift = bn_beta[c] - bn_mean[c] * scale;
#pragma unroll
    for (int v = 0; v < 8; ++v) {
      const int orow = permRow[w][lh * 8 + v];          // out[perm[n]] = z[n]
      const size_t off = ((size_t)b * kN + orow) * kC + c;
      out[off] = (acc[v] + wb) * scale + shift + x[off]; // residual = xg[n][c]
    }
  }
}

extern "C" void kernel_launch(void* const* d_in, const int* in_sizes, int n_in,
                              void* d_out, int out_size, void* d_ws, size_t ws_size,
                              hipStream_t stream) {
  const float* x        = (const float*)d_in[0];
  const int*   perm     = (const int*)d_in[1];
  // d_in[2] (restored_order) is unused: we scatter through perm instead.
  const float* g_w      = (const float*)d_in[3];
  const float* g_b      = (const float*)d_in[4];
  const float* theta_w  = (const float*)d_in[5];
  const float* theta_b  = (const float*)d_in[6];
  const float* phi_w    = (const float*)d_in[7];
  const float* phi_b    = (const float*)d_in[8];
  const float* W_w      = (const float*)d_in[9];
  const float* W_b      = (const float*)d_in[10];
  const float* bn_gamma = (const float*)d_in[11];
  const float* bn_beta  = (const float*)d_in[12];
  const float* bn_mean  = (const float*)d_in[13];
  const float* bn_var   = (const float*)d_in[14];
  float* out = (float*)d_out;

  // Workspace layout (f16): theta16 | phi16 | gT16 | tw16 | gw16 | pw16 | Ww16
  _Float16* theta16 = (_Float16*)d_ws;                      // B*N*D
  _Float16* phi16   = theta16 + (size_t)kB * kN * kD;       // B*M*D
  _Float16* gT16    = phi16   + (size_t)kB * kM * kD;       // B*D*M (transposed)
  _Float16* tw16    = gT16    + (size_t)kB * kD * kM;       // D*C
  _Float16* gw16    = tw16    + (size_t)kD * kC;
  _Float16* pw16    = gw16    + (size_t)kD * kC;
  _Float16* Ww16    = pw16    + (size_t)kD * kC;            // C*D

  k_wcvt<<<(kD * kC + 255) / 256, 256, 0, stream>>>(
      theta_w, g_w, phi_w, W_w, tw16, gw16, pw16, Ww16);

  const int strips = kB * (kN / 16);   // 16384, two waves per block
  k_proj<<<strips / 2, 64, 0, stream>>>(
      x, perm, tw16, gw16, pw16, theta_b, g_b, phi_b, theta16, phi16, gT16);
  k_attn<<<strips / 2, 64, 0, stream>>>(
      x, perm, theta16, phi16, gT16, Ww16, W_b,
      bn_gamma, bn_beta, bn_mean, bn_var, out);
}
